// MoETransformerWithReasoning_79061757984842
// MI455X (gfx1250) — matrix operand order repaired and checked by reading, not compiled, not run on hardware
//
#include <hip/hip_runtime.h>
#include <hip/hip_bf16.h>

// ---------------- problem constants (match reference) ----------------
#define B_TOK  2048
#define D_IN   2048
#define H_DIM  1024
#define L_NUM  3
#define E_NUM  8
#define C_CLS  1000
#define FF_DIM 4096

typedef __bf16 bf16_t;
typedef __attribute__((ext_vector_type(16))) __bf16 v16bf;
typedef __attribute__((ext_vector_type(8)))  float  v8f;

// ---------------- helpers ----------------
__device__ __forceinline__ unsigned short bfbits(float f) {
  unsigned u = __builtin_bit_cast(unsigned, f);
  u += 0x7FFFu + ((u >> 16) & 1u);            // round-to-nearest-even
  return (unsigned short)(u >> 16);
}
__device__ __forceinline__ bf16_t f2bf(float f) {
  unsigned short h = bfbits(f);
  return __builtin_bit_cast(bf16_t, h);
}
// interleave two packed-bf16 words (row k, row k+1) into two (k,k+1) pair words
// single v_perm_b32 each: o0=[lo0,lo1,hi0,hi1], o1=[lo2,lo3,hi2,hi3]
__device__ __forceinline__ void ilv2(unsigned lo, unsigned hi, unsigned& o0, unsigned& o1) {
  o0 = __builtin_amdgcn_perm(hi, lo, 0x05040100u);
  o1 = __builtin_amdgcn_perm(hi, lo, 0x07060302u);
}

// ---------------- fp32 -> bf16 converters ----------------
__global__ __launch_bounds__(256) void cvt_f32_bf16_kernel(
    const float* __restrict__ s, bf16_t* __restrict__ d, long n4) {
  long i = (long)blockIdx.x * 256 + threadIdx.x;
  if (i >= n4) return;
  float4 v = ((const float4*)s)[i];
  ushort4 o;
  o.x = bfbits(v.x); o.y = bfbits(v.y); o.z = bfbits(v.z); o.w = bfbits(v.w);
  ((ushort4*)d)[i] = o;
}

// strided slice convert: d[r*cols+c] = s[r*sld + soff + c]  (for V block of qkv_w)
__global__ __launch_bounds__(256) void cvt_slice_kernel(
    const float* __restrict__ s, bf16_t* __restrict__ d,
    int rows, int cols, int sld, int soff) {
  int n4 = rows * (cols >> 2);
  int i = blockIdx.x * 256 + threadIdx.x;
  if (i >= n4) return;
  int c4 = cols >> 2;
  int r = i / c4, c = (i - r * c4) * 4;
  float4 v = *(const float4*)(s + (long)r * sld + soff + c);
  ushort4 o;
  o.x = bfbits(v.x); o.y = bfbits(v.y); o.z = bfbits(v.z); o.w = bfbits(v.w);
  *(ushort4*)(d + (long)r * cols + c) = o;
}

// ---------------- WMMA GEMM ----------------
// C[M,N] = A[M,K](bf16) @ B[K,N](bf16) (+bias) (+gelu) -> f32/bf16/scatter-add
#define BM 128
#define BN 128
#define BK 32

template<bool GATHER_A, bool GELU, bool SCATTER, bool NGUARD>
__global__ __launch_bounds__(256) void gemm_bf16_wmma(
    const bf16_t* __restrict__ A, int lda,
    const bf16_t* __restrict__ Bw, int ldb,
    const float* __restrict__ bias,
    float* __restrict__ outF, bf16_t* __restrict__ outB, int ldc,
    int M, int N, int K,
    const int* __restrict__ gidx, const int* __restrict__ countPtr,
    float* __restrict__ scatterOut, const float* __restrict__ cwAll, int expertId) {
  const int tid  = threadIdx.x;
  const int lane = tid & 31;
  const int wv   = tid >> 5;         // 8 waves
  const int waveRow = wv >> 2;       // 0..1  (64 rows each)
  const int waveCol = wv & 3;        // 0..3  (32 cols each)
  const int lane16  = lane & 15;
  const int half    = lane >> 4;
  const int mBase = blockIdx.y * BM;
  const int nBase = blockIdx.x * BN;

  int mLimit = M;
  if (GATHER_A || SCATTER) {
    mLimit = *countPtr;
    if (mBase >= mLimit) return;     // uniform early-exit for empty tiles
  }

  // double-buffered LDS; A row-major (+8 pad), B as (k,k+1) bf16-pair words
  __shared__ bf16_t  As[2][BM][BK + 8];
  __shared__ unsigned Bp[2][BK / 2][BN + 4];

  // global->LDS mapping
  const int aC  = (tid & 3) * 8;     // 0/8/16/24
  const int aR0 = tid >> 2;          // 0..63  (second chunk +64)
  const int pr  = tid >> 4;          // B pair-row 0..15  (k = 2*pr, 2*pr+1)
  const int bC  = (tid & 15) * 8;    // col chunk 0..120
  const bool bIn = !NGUARD || (nBase + bC) < N;  // N%8==0 in all our cases

  long aOff0, aOff1;
  {
    int rr0 = mBase + aR0, rr1 = mBase + aR0 + 64;
    int t0 = rr0, t1 = rr1;
    if (GATHER_A) {
      t0 = (rr0 < mLimit) ? gidx[rr0] : 0;
      t1 = (rr1 < mLimit) ? gidx[rr1] : 0;
    }
    aOff0 = (long)t0 * lda;
    aOff1 = (long)t1 * lda;
  }
  const bf16_t* aP0 = A + aOff0 + aC;
  const bf16_t* aP1 = A + aOff1 + aC;
  const bf16_t* bP0 = Bw + (long)(2 * pr) * ldb + nBase + bC;
  const bf16_t* bP1 = bP0 + ldb;
  const long bStep = (long)BK * ldb;

  uint4 ra0, ra1, rb0, rb1;
  auto gload = [&]() {
    ra0 = *(const uint4*)aP0;
    ra1 = *(const uint4*)aP1;
    if (bIn) {
      rb0 = *(const uint4*)bP0;
      rb1 = *(const uint4*)bP1;
    } else {
      rb0 = uint4{0, 0, 0, 0};
      rb1 = uint4{0, 0, 0, 0};
    }
    aP0 += BK; aP1 += BK; bP0 += bStep; bP1 += bStep;
  };
  auto sstore = [&](int p) {
    *(uint4*)&As[p][aR0][aC]      = ra0;
    *(uint4*)&As[p][aR0 + 64][aC] = ra1;
    uint4 o0, o1;
    ilv2(rb0.x, rb1.x, o0.x, o0.y);
    ilv2(rb0.y, rb1.y, o0.z, o0.w);
    ilv2(rb0.z, rb1.z, o1.x, o1.y);
    ilv2(rb0.w, rb1.w, o1.z, o1.w);
    *(uint4*)&Bp[p][pr][bC]     = o0;
    *(uint4*)&Bp[p][pr][bC + 4] = o1;
  };

  v8f acc[4][2];
#pragma unroll
  for (int mi = 0; mi < 4; ++mi)
#pragma unroll
    for (int ni = 0; ni < 2; ++ni)
#pragma unroll
      for (int r = 0; r < 8; ++r) acc[mi][ni][r] = 0.0f;

  gload();
  sstore(0);
  __syncthreads();

  const int nk = K / BK;
  for (int i = 0; i < nk; ++i) {
    const int p = i & 1;
    const bool hasNext = (i + 1) < nk;
    if (hasNext) {
      gload();                      // issue next tile's global loads, overlap compute
      __builtin_prefetch(aP0, 0, 0);  // warm L2 for tile after next
      __builtin_prefetch(bP0, 0, 0);
    }

    // fragments in contiguous k order (k = 16*half + s) for both A and B
    union { v16bf v; unsigned u[8]; } afr[4];
#pragma unroll
    for (int mi = 0; mi < 4; ++mi) {
      int row = waveRow * 64 + mi * 16 + lane16;
#pragma unroll
      for (int q = 0; q < 8; ++q) {
        int k0 = half * 16 + q * 2;
        afr[mi].u[q] = *(const unsigned*)&As[p][row][k0];
      }
    }
    union { v16bf v; unsigned u[8]; } bfr[2];
#pragma unroll
    for (int ni = 0; ni < 2; ++ni) {
      int col = waveCol * 32 + ni * 16 + lane16;
#pragma unroll
      for (int q = 0; q < 8; ++q)
        bfr[ni].u[q] = Bp[p][half * 8 + q][col];
    }

#pragma unroll
    for (int mi = 0; mi < 4; ++mi)
#pragma unroll
      for (int ni = 0; ni < 2; ++ni)
        acc[mi][ni] = __builtin_amdgcn_wmma_f32_16x16x32_bf16(
            false, afr[mi].v, false, bfr[ni].v, (short)0, acc[mi][ni],
            false, false);

    if (hasNext) sstore(1 - p);     // write alternate buffer (no reader this iter)
    __syncthreads();
  }

  // epilogue (C layout: VGPR r -> M=r (lanes 0-15) / M=8+r (lanes 16-31))
#pragma unroll
  for (int mi = 0; mi < 4; ++mi) {
#pragma unroll
    for (int ni = 0; ni < 2; ++ni) {
#pragma unroll
      for (int r = 0; r < 8; ++r) {
        int row = mBase + waveRow * 64 + mi * 16 + half * 8 + r;
        int col = nBase + waveCol * 32 + ni * 16 + lane16;
        if (NGUARD && col >= N) continue;
        float v = acc[mi][ni][r];
        if (bias) v += bias[col];
        if (GELU) v = 0.5f * v * (1.0f + erff(v * 0.70710678118654752f));
        if (SCATTER) {
          if (row < mLimit) {
            int tok = gidx[row];
            float w = cwAll[tok * E_NUM + expertId];
            atomicAdd(&scatterOut[(long)tok * ldc + col], w * v);
          }
        } else {
          if (GATHER_A && row >= mLimit) continue;
          if (outF) outF[(long)row * ldc + col] = v;
          if (outB) outB[(long)row * ldc + col] = f2bf(v);
        }
      }
    }
  }
}

// ---------------- residual + layernorm ----------------
__global__ __launch_bounds__(256) void add_ln_kernel(
    const float* __restrict__ hf, const float* __restrict__ af,
    const float* __restrict__ g, const float* __restrict__ b,
    float* __restrict__ of, bf16_t* __restrict__ ob) {
  int row = blockIdx.x, tid = threadIdx.x;
  const float* hr = hf + (long)row * H_DIM;
  const float* ar = af + (long)row * H_DIM;
  float x[4], s = 0.f, ss = 0.f;
#pragma unroll
  for (int j = 0; j < 4; ++j) {
    int c = tid + j * 256;
    x[j] = hr[c] + ar[c];
    s += x[j]; ss += x[j] * x[j];
  }
  __shared__ float r1[256], r2[256];
  r1[tid] = s; r2[tid] = ss;
  __syncthreads();
  for (int o = 128; o > 0; o >>= 1) {
    if (tid < o) { r1[tid] += r1[tid + o]; r2[tid] += r2[tid + o]; }
    __syncthreads();
  }
  float mean = r1[0] * (1.0f / H_DIM);
  float var  = r2[0] * (1.0f / H_DIM) - mean * mean;
  float inv  = rsqrtf(var + 1e-5f);
#pragma unroll
  for (int j = 0; j < 4; ++j) {
    int c = tid + j * 256;
    float y = (x[j] - mean) * inv * g[c] + b[c];
    of[(long)row * H_DIM + c] = y;
    ob[(long)row * H_DIM + c] = f2bf(y);
  }
}

// ---------------- gate: softmax + top-2 renorm -> dense combine weights ----------------
__global__ __launch_bounds__(256) void gate_topk_kernel(
    const float* __restrict__ hf, const float* __restrict__ gw,
    const float* __restrict__ gb, float* __restrict__ cw) {
  int tok = blockIdx.x, tid = threadIdx.x;
  int lane = tid & 31, e = tid >> 5;      // one wave32 per expert (E_NUM==8)
  const float* hr = hf + (long)tok * H_DIM;
  float s = 0.f;
  for (int i = lane; i < H_DIM; i += 32) s += hr[i] * gw[i * E_NUM + e];
  for (int o = 16; o > 0; o >>= 1) s += __shfl_xor(s, o, 32);
  __shared__ float gl[E_NUM];
  if (lane == 0) gl[e] = s + gb[e];
  __syncthreads();
  if (tid == 0) {
    float p[E_NUM], mx = gl[0];
    for (int i = 1; i < E_NUM; ++i) mx = fmaxf(mx, gl[i]);
    float sum = 0.f;
    for (int i = 0; i < E_NUM; ++i) { p[i] = expf(gl[i] - mx); sum += p[i]; }
    for (int i = 0; i < E_NUM; ++i) p[i] /= sum;
    int i0 = 0;
    for (int i = 1; i < E_NUM; ++i) if (p[i] > p[i0]) i0 = i;
    int i1 = (i0 == 0) ? 1 : 0;
    for (int i = 0; i < E_NUM; ++i) if (i != i0 && p[i] > p[i1]) i1 = i;
    float s2 = p[i0] + p[i1];
    for (int i = 0; i < E_NUM; ++i) cw[tok * E_NUM + i] = 0.f;
    cw[tok * E_NUM + i0] = p[i0] / s2;
    cw[tok * E_NUM + i1] = p[i1] / s2;
  }
}

// ---------------- build per-expert token lists ----------------
__global__ __launch_bounds__(256) void route_kernel(
    const float* __restrict__ cw, int* __restrict__ idxBuf, int* __restrict__ cursors) {
  int t = blockIdx.x * 256 + threadIdx.x;
  if (t >= B_TOK) return;
  for (int e = 0; e < E_NUM; ++e)
    if (cw[t * E_NUM + e] > 0.f) {
      int pos = atomicAdd(&cursors[e], 1);
      idxBuf[e * B_TOK + pos] = t;
    }
}

// ---------------- host orchestration ----------------
extern "C" void kernel_launch(void* const* d_in, const int* in_sizes, int n_in,
                              void* d_out, int out_size, void* d_ws, size_t ws_size,
                              hipStream_t stream) {
  (void)in_sizes; (void)n_in; (void)out_size; (void)ws_size;
  const float* x          = (const float*)d_in[0];
  const float* proj_w     = (const float*)d_in[1];
  const float* proj_b     = (const float*)d_in[2];
  const float* qkv_w      = (const float*)d_in[3];
  const float* qkv_b      = (const float*)d_in[4];
  const float* attn_out_w = (const float*)d_in[5];
  const float* attn_out_b = (const float*)d_in[6];
  const float* ln_g       = (const float*)d_in[7];
  const float* ln_b       = (const float*)d_in[8];
  const float* gate_w     = (const float*)d_in[9];
  const float* gate_b     = (const float*)d_in[10];
  const float* e_w1       = (const float*)d_in[11];
  const float* e_b1       = (const float*)d_in[12];
  const float* e_w2       = (const float*)d_in[13];
  const float* e_b2       = (const float*)d_in[14];
  const float* cls_w      = (const float*)d_in[15];
  const float* cls_b      = (const float*)d_in[16];
  float* logits = (float*)d_out;

  // ---- carve workspace ----
  size_t off = 0;
  auto carve = [&](size_t bytes) {
    off = (off + 255) & ~(size_t)255;
    void* p = (char*)d_ws + off;
    off += bytes;
    return p;
  };
  bf16_t* wx    = (bf16_t*)carve((size_t)B_TOK * D_IN * 2);
  bf16_t* wproj = (bf16_t*)carve((size_t)D_IN * H_DIM * 2);
  bf16_t* wvv   = (bf16_t*)carve((size_t)H_DIM * H_DIM * 2);
  bf16_t* wo    = (bf16_t*)carve((size_t)H_DIM * H_DIM * 2);
  bf16_t* wcls  = (bf16_t*)carve((size_t)H_DIM * C_CLS * 2);
  bf16_t* ew1   = (bf16_t*)carve((size_t)E_NUM * H_DIM * FF_DIM * 2);
  bf16_t* ew2   = (bf16_t*)carve((size_t)E_NUM * H_DIM * FF_DIM * 2);
  float*  bufA  = (float*)carve((size_t)B_TOK * H_DIM * 4);
  float*  bufB  = (float*)carve((size_t)B_TOK * H_DIM * 4);
  bf16_t* hb    = (bf16_t*)carve((size_t)B_TOK * H_DIM * 2);
  bf16_t* vb    = (bf16_t*)carve((size_t)B_TOK * H_DIM * 2);
  float*  af    = (float*)carve((size_t)B_TOK * H_DIM * 4);
  bf16_t* t1    = (bf16_t*)carve((size_t)B_TOK * FF_DIM * 2);
  float*  cw    = (float*)carve((size_t)B_TOK * E_NUM * 4);
  int*    idx   = (int*)carve((size_t)E_NUM * B_TOK * 4);
  int*    cur   = (int*)carve((size_t)E_NUM * 4);

  auto cvt = [&](const float* s, bf16_t* d, size_t n) {
    long n4 = (long)(n / 4);
    cvt_f32_bf16_kernel<<<(unsigned)((n4 + 255) / 256), 256, 0, stream>>>(s, d, n4);
  };

  const dim3 blk(256);
  const dim3 gH(H_DIM / BN, B_TOK / BM);    // (8,16)
  const dim3 gFF(FF_DIM / BN, B_TOK / BM);  // (32,16)
  const dim3 gC((C_CLS + BN - 1) / BN, B_TOK / BM);

  // ---- one-shot conversions ----
  cvt(x, wx, (size_t)B_TOK * D_IN);
  cvt(proj_w, wproj, (size_t)D_IN * H_DIM);
  cvt(cls_w, wcls, (size_t)H_DIM * C_CLS);

  // ---- input projection: hf,hb = x @ proj_w + proj_b ----
  float* hf = bufA;
  float* macc = bufB;
  gemm_bf16_wmma<false, false, false, false><<<gH, blk, 0, stream>>>(
      wx, D_IN, wproj, H_DIM, proj_b, hf, hb, H_DIM,
      B_TOK, H_DIM, D_IN, nullptr, nullptr, nullptr, nullptr, 0);

  for (int l = 0; l < L_NUM; ++l) {
    // per-layer weight conversion (bf16 expert weights stay L2-resident)
    cvt_slice_kernel<<<(H_DIM * H_DIM / 4 + 255) / 256, 256, 0, stream>>>(
        qkv_w + (size_t)l * H_DIM * 3 * H_DIM, wvv, H_DIM, H_DIM, 3 * H_DIM, 2 * H_DIM);
    cvt(attn_out_w + (size_t)l * H_DIM * H_DIM, wo, (size_t)H_DIM * H_DIM);
    cvt(e_w1 + (size_t)l * E_NUM * H_DIM * FF_DIM, ew1, (size_t)E_NUM * H_DIM * FF_DIM);
    cvt(e_w2 + (size_t)l * E_NUM * H_DIM * FF_DIM, ew2, (size_t)E_NUM * H_DIM * FF_DIM);

    // S==1 attention collapses to v = h@Wv+bv ; a = v@Wo+bo
    gemm_bf16_wmma<false, false, false, false><<<gH, blk, 0, stream>>>(
        hb, H_DIM, wvv, H_DIM, qkv_b + (size_t)l * 3 * H_DIM + 2 * H_DIM,
        nullptr, vb, H_DIM, B_TOK, H_DIM, H_DIM,
        nullptr, nullptr, nullptr, nullptr, 0);
    gemm_bf16_wmma<false, false, false, false><<<gH, blk, 0, stream>>>(
        vb, H_DIM, wo, H_DIM, attn_out_b + (size_t)l * H_DIM,
        af, nullptr, H_DIM, B_TOK, H_DIM, H_DIM,
        nullptr, nullptr, nullptr, nullptr, 0);

    // h = LN(h + a)
    add_ln_kernel<<<B_TOK, blk, 0, stream>>>(
        hf, af, ln_g + (size_t)l * H_DIM, ln_b + (size_t)l * H_DIM, hf, hb);

    // gating + routing
    gate_topk_kernel<<<B_TOK, blk, 0, stream>>>(
        hf, gate_w + (size_t)l * H_DIM * E_NUM, gate_b + (size_t)l * E_NUM, cw);
    hipMemsetAsync(cur, 0, E_NUM * sizeof(int), stream);
    hipMemsetAsync(macc, 0, (size_t)B_TOK * H_DIM * 4, stream);
    route_kernel<<<(B_TOK + 255) / 256, blk, 0, stream>>>(cw, idx, cur);

    // sparse expert FFNs (gathered GEMM -> GELU -> GEMM -> weighted scatter-add)
    for (int e = 0; e < E_NUM; ++e) {
      gemm_bf16_wmma<true, true, false, false><<<gFF, blk, 0, stream>>>(
          hb, H_DIM, ew1 + (size_t)e * H_DIM * FF_DIM, FF_DIM,
          e_b1 + ((size_t)l * E_NUM + e) * FF_DIM,
          nullptr, t1, FF_DIM, B_TOK, FF_DIM, H_DIM,
          idx + e * B_TOK, cur + e, nullptr, nullptr, 0);
      gemm_bf16_wmma<false, false, true, false><<<gH, blk, 0, stream>>>(
          t1, FF_DIM, ew2 + (size_t)e * H_DIM * FF_DIM, H_DIM,
          e_b2 + ((size_t)l * E_NUM + e) * H_DIM,
          nullptr, nullptr, H_DIM, B_TOK, H_DIM, FF_DIM,
          idx + e * B_TOK, cur + e, macc, cw, e);
    }

    // h <- moe output (ping-pong), refresh bf16 mirror
    float* tmp = hf; hf = macc; macc = tmp;
    cvt(hf, hb, (size_t)B_TOK * H_DIM);
  }

  // logits = h @ cls_w + cls_b   (mean over S=1 is identity; N=1000 guarded)
  gemm_bf16_wmma<false, false, false, true><<<gC, blk, 0, stream>>>(
      hb, H_DIM, wcls, C_CLS, cls_b, logits, nullptr, C_CLS,
      B_TOK, C_CLS, H_DIM, nullptr, nullptr, nullptr, nullptr, 0);
}
